// LogicalAttentionLayer_50964081934792
// MI455X (gfx1250) — matrix operand order repaired and checked by reading, not compiled, not run on hardware
//
#include <hip/hip_runtime.h>
#include <hip/hip_bf16.h>

// B=4, S=4096, HID=1024, NH=16, HD=64  ->  M = B*S = 16384 tokens.
#define MDIM 16384
#define NDIM 1024
#define KDIM 1024

typedef __attribute__((ext_vector_type(16))) __bf16 v16bf;
typedef __attribute__((ext_vector_type(8)))  __bf16 v8bf;
typedef __attribute__((ext_vector_type(8)))  float  v8f;
typedef __attribute__((ext_vector_type(4)))  int    v4i;

static __device__ __forceinline__ v16bf cat16(v8bf lo, v8bf hi) {
    return __builtin_shufflevector(lo, hi, 0,1,2,3,4,5,6,7,8,9,10,11,12,13,14,15);
}

// ---- CDNA5 async global->LDS copy (16 B), with portable fallback -----------
#if __has_builtin(__builtin_amdgcn_global_load_async_to_lds_b128)
#define HAVE_ASYNC_LDS 1
#else
#define HAVE_ASYNC_LDS 0
#endif

static __device__ __forceinline__ void cp16_g2l(__bf16* l, const __bf16* g) {
#if HAVE_ASYNC_LDS
    // signature (from hipcc diagnostic): (v4i AS1*, v4i AS3*, imm offset, imm cpol)
    __builtin_amdgcn_global_load_async_to_lds_b128(
        (__attribute__((address_space(1))) v4i*)g,
        (__attribute__((address_space(3))) v4i*)l, 0, 0);
#else
    *(v8bf*)l = *(const v8bf*)g;
#endif
}

static __device__ __forceinline__ void wait_async() {
#if HAVE_ASYNC_LDS
#if __has_builtin(__builtin_amdgcn_s_wait_asynccnt)
    __builtin_amdgcn_s_wait_asynccnt(0);
#else
    asm volatile("s_wait_asynccnt 0" ::: "memory");
#endif
#endif
}

// ---------------------------------------------------------------------------
// fp32 -> bf16 pre-conversion (done ONCE so GEMM staging is a pure DMA copy).
// n is a multiple of 2048; each thread converts 8 elements.
// ---------------------------------------------------------------------------
__global__ __launch_bounds__(256)
void conv_f32_bf16_k(const float* __restrict__ s, __bf16* __restrict__ d)
{
    size_t i = (size_t)blockIdx.x * 256 + threadIdx.x;   // unit of 8 elements
    const float4* s4 = (const float4*)s;
    float4 a = s4[i * 2 + 0];
    float4 b = s4[i * 2 + 1];
    v8bf o;
    o[0] = (__bf16)a.x; o[1] = (__bf16)a.y; o[2] = (__bf16)a.z; o[3] = (__bf16)a.w;
    o[4] = (__bf16)b.x; o[5] = (__bf16)b.y; o[6] = (__bf16)b.z; o[7] = (__bf16)b.w;
    ((v8bf*)d)[i] = o;
}

// ---------------------------------------------------------------------------
// Y[M,N] = A[M,K](bf16) @ W[N,K](bf16)^T + b[N](f32)
// Block: 128 threads = 4 wave32, block tile 128x128, K staged 64 wide,
// double-buffered LDS filled by async global->LDS DMA.
// Each wave owns a 64x64 tile = 4x4 chained v_wmma_f32_16x16x32_bf16
// accumulators -> 16 WMMA per 16 ds_load_b128 in the inner loop.
// ---------------------------------------------------------------------------
template<bool OUT_BF16>
__global__ __launch_bounds__(128)
void gemm_wmma_k(const __bf16* __restrict__ A, const __bf16* __restrict__ W,
                 const float* __restrict__ bias, void* Outp,
                 int M, int N, int K)
{
    // +8 half padding -> row stride 144 B (16B-aligned, bank-spread). 72 KB.
    __shared__ __bf16 As[2][128][72];
    __shared__ __bf16 Bs[2][128][72];

    const int tid  = threadIdx.x;
    const int lane = tid & 31;
    const int wv   = tid >> 5;          // 0..3
    const int wm   = (wv >> 1) * 64;    // wave row offset in block tile
    const int wn   = (wv & 1) * 64;     // wave col offset in block tile
    const int m0   = blockIdx.x * 128;
    const int n0   = blockIdx.y * 128;
    const int lr   = lane & 15;
    const int kh   = lane >> 4;
    const int NK   = K / 64;

    v8f c[4][4];
    #pragma unroll
    for (int i = 0; i < 4; ++i)
        #pragma unroll
        for (int j = 0; j < 4; ++j) { v8f z = {}; c[i][j] = z; }

    // stage one 128x64 A tile + 128x64 B tile into buffer `buf`
    auto stage = [&](int buf, int kt) {
        #pragma unroll
        for (int u = 0; u < 8; ++u) {               // 1024 16B units / 128 thr
            int lin = tid + u * 128;
            int r = lin >> 3, cu = lin & 7;
            cp16_g2l(&As[buf][r][cu * 8], A + (size_t)(m0 + r) * K + kt + cu * 8);
        }
        #pragma unroll
        for (int u = 0; u < 8; ++u) {
            int lin = tid + u * 128;
            int r = lin >> 3, cu = lin & 7;
            cp16_g2l(&Bs[buf][r][cu * 8], W + (size_t)(n0 + r) * K + kt + cu * 8);
        }
    };

    int cur = 0;
    stage(0, 0);
    for (int kt = 0; kt < NK; ++kt) {
        wait_async();                 // my share of buf `cur` has landed
        __syncthreads();              // everyone's share has landed
        if (kt + 1 < NK) stage(cur ^ 1, (kt + 1) * 64);   // overlap with math

        #pragma unroll
        for (int ks = 0; ks < 64; ks += 32) {
            v16bf a[4], b[4];
            #pragma unroll
            for (int i = 0; i < 4; ++i) {           // A frags (16x32)
                int mr = wm + i * 16 + lr;
                v8bf lo = *(const v8bf*)&As[cur][mr][ks + kh * 8];
                v8bf hi = *(const v8bf*)&As[cur][mr][ks + 16 + kh * 8];
                a[i] = cat16(lo, hi);
            }
            #pragma unroll
            for (int j = 0; j < 4; ++j) {           // B frags (32x16)
                int nr = wn + j * 16 + lr;
                v8bf lo = *(const v8bf*)&Bs[cur][nr][ks + kh * 8];
                v8bf hi = *(const v8bf*)&Bs[cur][nr][ks + 16 + kh * 8];
                b[j] = cat16(lo, hi);
            }
            #pragma unroll
            for (int i = 0; i < 4; ++i)
                #pragma unroll
                for (int j = 0; j < 4; ++j)
                    c[i][j] = __builtin_amdgcn_wmma_f32_16x16x32_bf16(
                        false, a[i], false, b[j], (short)0, c[i][j], false, false);
        }
        cur ^= 1;
    }

    // epilogue: D layout = lane col N (lr), 8 rows M = r + 8*kh
    #pragma unroll
    for (int i = 0; i < 4; ++i)
        #pragma unroll
        for (int j = 0; j < 4; ++j) {
            int n  = n0 + wn + j * 16 + lr;
            float bv = bias[n];
            int mb = m0 + wm + i * 16 + kh * 8;
            if constexpr (OUT_BF16) {
                __bf16* O = (__bf16*)Outp;
                #pragma unroll
                for (int r = 0; r < 8; ++r)
                    O[(size_t)(mb + r) * N + n] = (__bf16)(c[i][j][r] + bv);
            } else {
                float* O = (float*)Outp;
                #pragma unroll
                for (int r = 0; r < 8; ++r)
                    O[(size_t)(mb + r) * N + n] = c[i][j][r] + bv;
            }
        }
}

// ---------------------------------------------------------------------------
// Per-token head-mixing attention: one wave per token.
//   scores[16,16] = (q[16,64] . k[16,64]^T) / 8   -> 2x chained WMMA (K=64)
//   softmax over head axis (16-lane shfl butterfly per D-layout row)
//   out[16,64]   = attn[16,16] @ v[16,64]         -> 4x WMMA, K padded 16->32
// v is staged TRANSPOSED in LDS so B-fragments are contiguous reads.
// `out` aliases `q` (in-place); q is fully consumed before the stores.
// ---------------------------------------------------------------------------
__global__ __launch_bounds__(256)
void attn_headmix_k(const __bf16* q, const __bf16* __restrict__ k,
                    const __bf16* __restrict__ v, __bf16* out)
{
    __shared__ __bf16 attn_s[8][16][16];   // 4 KB
    __shared__ __bf16 vT_s[8][64][16];     // 16 KB : vT[d][g] = v[g][d]

    const int tid  = threadIdx.x;
    const int lane = tid & 31;
    const int w    = tid >> 5;
    const size_t t = (size_t)blockIdx.x * 8 + w;   // token id, exact fit
    const __bf16* qt = q + t * 1024;
    const __bf16* kt = k + t * 1024;
    const __bf16* vt = v + t * 1024;
    const int lr = lane & 15;
    const int kh = lane >> 4;

    // stage v transposed: lane covers row g = lane/2, d-half = (lane&1)*32
    {
        int g  = lane >> 1;
        int dh = (lane & 1) * 32;
        #pragma unroll
        for (int j = 0; j < 4; ++j) {
            v8bf vv = *(const v8bf*)(vt + g * 64 + dh + j * 8);
            #pragma unroll
            for (int e = 0; e < 8; ++e)
                vT_s[w][dh + j * 8 + e][g] = vv[e];
        }
    }

    // scores = q . k^T over d=64 : two chained K=32 WMMAs
    v8f c = {};
    #pragma unroll
    for (int jk = 0; jk < 2; ++jk) {
        int db = jk * 32;
        v8bf alo = *(const v8bf*)(qt + lr * 64 + db + kh * 8);
        v8bf ahi = *(const v8bf*)(qt + lr * 64 + db + 16 + kh * 8);
        v16bf a  = cat16(alo, ahi);
        v8bf blo = *(const v8bf*)(kt + lr * 64 + db + kh * 8);
        v8bf bhi = *(const v8bf*)(kt + lr * 64 + db + 16 + kh * 8);
        v16bf b  = cat16(blo, bhi);
        c = __builtin_amdgcn_wmma_f32_16x16x32_bf16(
                false, a, false, b, (short)0, c, false, false);
    }

    // softmax over cols (head axis g): each D row spans 16 lanes of one VGPR
    #pragma unroll
    for (int r = 0; r < 8; ++r) {
        float xs = c[r] * 0.125f;                  // 1/sqrt(64)
        float m = xs;
        m = fmaxf(m, __shfl_xor(m, 1, 16));
        m = fmaxf(m, __shfl_xor(m, 2, 16));
        m = fmaxf(m, __shfl_xor(m, 4, 16));
        m = fmaxf(m, __shfl_xor(m, 8, 16));
        float e = __expf(xs - m);
        float s = e;
        s += __shfl_xor(s, 1, 16);
        s += __shfl_xor(s, 2, 16);
        s += __shfl_xor(s, 4, 16);
        s += __shfl_xor(s, 8, 16);
        attn_s[w][r + kh * 8][lr] = (__bf16)(e / s);
    }

    // out = attn @ v with K padded 16 -> 32 (upper half zeros).
    v8bf z8;
    #pragma unroll
    for (int e = 0; e < 8; ++e) z8[e] = (__bf16)0.0f;
    const v16bf zfrag = cat16(z8, z8);

    // A frag: elems 0..7 = attn[row lr][kh*8 .. kh*8+7], elems 8..15 = 0
    v8bf alo = *(const v8bf*)&attn_s[w][lr][kh * 8];
    v16bf a  = cat16(alo, z8);

    #pragma unroll
    for (int dt = 0; dt < 4; ++dt) {
        // B frag: elems 0..7 = vT[dt*16+lr][kh*8 .. kh*8+7], elems 8..15 = 0
        v8bf blo = *(const v8bf*)&vT_s[w][dt * 16 + lr][kh * 8];
        v8bf bhi = *(const v8bf*)&vT_s[w][dt * 16 + lr][8];
        v16bf b  = kh ? cat16(blo, z8) : cat16(blo, z8);
        v8f z = {};
        v8f o = __builtin_amdgcn_wmma_f32_16x16x32_bf16(
                    false, a, false, b, (short)0, z, false, false);
        #pragma unroll
        for (int r = 0; r < 8; ++r)
            out[t * 1024 + (size_t)(r + 8 * kh) * 64 + dt * 16 + lr] = (__bf16)o[r];
    }
}

// ---------------------------------------------------------------------------
extern "C" void kernel_launch(void* const* d_in, const int* in_sizes, int n_in,
                              void* d_out, int out_size, void* d_ws, size_t ws_size,
                              hipStream_t stream)
{
    const float* x  = (const float*)d_in[0];
    const float* Wq = (const float*)d_in[1];
    const float* bq = (const float*)d_in[2];
    const float* Wk = (const float*)d_in[3];
    const float* bk = (const float*)d_in[4];
    const float* Wv = (const float*)d_in[5];
    const float* bv = (const float*)d_in[6];
    const float* Wo = (const float*)d_in[7];
    const float* bo = (const float*)d_in[8];

    const size_t tok  = (size_t)MDIM * NDIM;   // 16384*1024 elements
    const size_t wsz  = (size_t)NDIM * KDIM;   // 1024*1024 elements
    __bf16* xbf = (__bf16*)d_ws;
    __bf16* wqb = xbf + tok;                   // later reused for Wo
    __bf16* wkb = wqb + wsz;
    __bf16* wvb = wkb + wsz;
    __bf16* qws = wvb + wsz;                   // also reused as attn output
    __bf16* kws = qws + tok;
    __bf16* vws = kws + tok;

    // one-time fp32 -> bf16 conversion (avoids 8x redundant in-GEMM converts)
    conv_f32_bf16_k<<<tok / 2048, 256, 0, stream>>>(x,  xbf);
    conv_f32_bf16_k<<<wsz / 2048, 256, 0, stream>>>(Wq, wqb);
    conv_f32_bf16_k<<<wsz / 2048, 256, 0, stream>>>(Wk, wkb);
    conv_f32_bf16_k<<<wsz / 2048, 256, 0, stream>>>(Wv, wvb);

    dim3 gg(MDIM / 128, NDIM / 128);           // (128, 8)
    dim3 bb(128);
    gemm_wmma_k<true><<<gg, bb, 0, stream>>>(xbf, wqb, bq, qws, MDIM, NDIM, KDIM);
    gemm_wmma_k<true><<<gg, bb, 0, stream>>>(xbf, wkb, bk, kws, MDIM, NDIM, KDIM);
    gemm_wmma_k<true><<<gg, bb, 0, stream>>>(xbf, wvb, bv, vws, MDIM, NDIM, KDIM);

    // Wq's bf16 slot is dead now -> reuse it for Wo (stream-ordered)
    conv_f32_bf16_k<<<wsz / 2048, 256, 0, stream>>>(Wo, wqb);

    attn_headmix_k<<<MDIM / 8, 256, 0, stream>>>(qws, kws, vws, qws);

    gemm_wmma_k<false><<<gg, bb, 0, stream>>>(qws, wqb, bo, (float*)d_out,
                                              MDIM, NDIM, KDIM);
}
